// VQVAELayer_80229989089999
// MI455X (gfx1250) — matrix operand order repaired and checked by reading, not compiled, not run on hardware
//
#include <hip/hip_runtime.h>
#include <hip/hip_bf16.h>

// ---------------------------------------------------------------------------
// VQ-VAE layer for MI455X (gfx1250, wave32).
// Distance GEMM + argmin fused on v_wmma_f32_16x16x32_bf16 (bf16 hi/lo split,
// 3 WMMAs per tile ~ fp32-accurate dot). Codebook panels are staged into LDS
// with CDNA5 async global->LDS loads, double-buffered across the code loop.
// ---------------------------------------------------------------------------

typedef __attribute__((ext_vector_type(16))) __bf16   v16bf;
typedef __attribute__((ext_vector_type(8)))  float    v8f;
typedef __attribute__((ext_vector_type(4)))  unsigned u32x4;
typedef __attribute__((ext_vector_type(4)))  int      i32x4;

#define N_EMBS   1024
#define EMB_DIM  256
#define BATCH    32
#define HH       32
#define WW       32
#define NROWS    (BATCH * HH * WW)      // 32768 spatial positions
#define BETA     0.25f
#define DECAY    0.99f
#define EPSV     1e-5f

static __device__ __forceinline__ unsigned short f2bf(float f) {
    unsigned u = __float_as_uint(f);
    unsigned r = u + 0x7FFFu + ((u >> 16) & 1u);   // round-to-nearest-even
    return (unsigned short)(r >> 16);
}
static __device__ __forceinline__ float bf2f(unsigned short s) {
    return __uint_as_float(((unsigned)s) << 16);
}

// ---- CDNA5 async global->LDS copy of one 16B chunk -------------------------
static __device__ __forceinline__ void async_copy16(unsigned short* lds,
                                                    const unsigned short* g) {
#if __has_builtin(__builtin_amdgcn_global_load_async_to_lds_b128)
    typedef __attribute__((address_space(1))) i32x4 gi4;   // global v4i
    typedef __attribute__((address_space(3))) i32x4 li4;   // LDS v4i
    __builtin_amdgcn_global_load_async_to_lds_b128((gi4*)g, (li4*)lds, 0, 0);
#else
    // GLOBAL_LOAD_ASYNC_TO_LDS_B128: VDST = LDS byte address, VADDR = 64b addr
    unsigned ldsoff = (unsigned)(size_t)lds;
    unsigned long long ga = (unsigned long long)(size_t)g;
    asm volatile("global_load_async_to_lds_b128 %0, %1, off"
                 :: "v"(ldsoff), "v"(ga) : "memory");
#endif
}
static __device__ __forceinline__ void wait_async0() {
#if __has_builtin(__builtin_amdgcn_s_wait_asynccnt)
    __builtin_amdgcn_s_wait_asynccnt(0);
#else
    asm volatile("s_wait_asynccnt 0x0" ::: "memory");
#endif
}

// ---------------------------------------------------------------------------
// Kernel 1: emb_weight (1024x256 fp32) -> bf16 hi/lo + per-code squared norm.
// ---------------------------------------------------------------------------
__global__ void k_prep_emb(const float* __restrict__ emb,
                           unsigned short* __restrict__ ehi,
                           unsigned short* __restrict__ elo,
                           float* __restrict__ e2) {
    int k = blockIdx.x, d = threadIdx.x;
    float w = emb[(size_t)k * EMB_DIM + d];
    unsigned short hi = f2bf(w);
    ehi[(size_t)k * EMB_DIM + d] = hi;
    elo[(size_t)k * EMB_DIM + d] = f2bf(w - bf2f(hi));
    __shared__ float red[EMB_DIM];
    red[d] = w * w;
    __syncthreads();
    for (int s = EMB_DIM / 2; s > 0; s >>= 1) {
        if (d < s) red[d] += red[d + s];
        __syncthreads();
    }
    if (d == 0) e2[k] = red[0];
}

// ---------------------------------------------------------------------------
// Kernel 2: x (B,D,H,W) fp32 -> flat (N,D) bf16 hi/lo via LDS-tiled transpose.
// grid = B*H*(D/32) blocks; block = 256 threads (32 w x 8).
// ---------------------------------------------------------------------------
__global__ void k_prep_x(const float* __restrict__ x,
                         unsigned short* __restrict__ xhi,
                         unsigned short* __restrict__ xlo) {
    int blk = blockIdx.x;
    int dt = blk & 7;            // D tile (8 tiles of 32)
    int h  = (blk >> 3) & 31;
    int b  = blk >> 8;
    int tx = threadIdx.x & 31;   // fast index (w on load, d on store)
    int ty = threadIdx.x >> 5;   // 0..7
    __shared__ float tile[32][33];
    #pragma unroll
    for (int i = 0; i < 4; i++) {
        int d = dt * 32 + ty + i * 8;
        tile[ty + i * 8][tx] = x[(((size_t)b * EMB_DIM + d) * HH + h) * WW + tx];
    }
    __syncthreads();
    #pragma unroll
    for (int i = 0; i < 4; i++) {
        int w = ty + i * 8;
        size_t n = (size_t)b * (HH * WW) + h * WW + w;
        float v = tile[tx][w];
        unsigned short hi = f2bf(v);
        xhi[n * EMB_DIM + dt * 32 + tx] = hi;
        xlo[n * EMB_DIM + dt * 32 + tx] = f2bf(v - bf2f(hi));
    }
}

// ---------------------------------------------------------------------------
// Kernel 3: fused distance-GEMM + argmin, B panels staged via async->LDS.
// score(n,k) = ||e_k||^2 - 2 * <x_n, e_k>  (row norm constant -> drops out)
//
// Block = 8 waves; wave owns a 16-row M tile. All 8 waves share each B panel:
// panel ct = codes [16ct, 16ct+16) x 256 dims, bf16 hi+lo = 8KB+8KB, stored
// contiguously in global -> straight async B128 copy, double-buffered in LDS.
//
// Fragment addressing (16-bit WMMA lane layouts, ISA 7.12.2):
//   A (16x32): row = mBase + (lane&15); per-lane K = {kb+sel*8..+7, kb+16+sel*8..+7}
//   B (32x16): col = code = cBase + (lane&15); per-lane K = {kb+sel*8..+7, kb+16+sel*8..+7}
//   C: lane holds rows {r + 8*sel}, column (lane&15).
// ---------------------------------------------------------------------------
#define PANEL (16 * EMB_DIM)   // 4096 bf16 elements = 8KB

static __device__ __forceinline__ void stage_panel(const unsigned short* gh,
                                                   const unsigned short* gl,
                                                   unsigned short* lh,
                                                   unsigned short* ll, int t) {
    // 8KB per array = 512 x 16B chunks; 256 threads move 2 chunks each.
    async_copy16(lh + (size_t)t * 8,         gh + (size_t)t * 8);
    async_copy16(lh + (size_t)(t + 256) * 8, gh + (size_t)(t + 256) * 8);
    async_copy16(ll + (size_t)t * 8,         gl + (size_t)t * 8);
    async_copy16(ll + (size_t)(t + 256) * 8, gl + (size_t)(t + 256) * 8);
}

__global__ void __launch_bounds__(256)
k_argmin(const unsigned short* __restrict__ xhi,
         const unsigned short* __restrict__ xlo,
         const unsigned short* __restrict__ ehi,
         const unsigned short* __restrict__ elo,
         const float* __restrict__ e2,
         int* __restrict__ bestIdx) {
    __shared__ unsigned short sbhi[2][PANEL];   // 16KB
    __shared__ unsigned short sblo[2][PANEL];   // 16KB
    int t    = threadIdx.x;
    int lane = t & 31;
    int wv   = t >> 5;                    // 8 waves per block
    int sel  = lane >> 4;                 // 0 or 1 (half-wave)
    int col  = lane & 15;
    int mBase = blockIdx.x * 128 + wv * 16;
    const unsigned short* arow_hi = xhi + (size_t)(mBase + col) * EMB_DIM;
    const unsigned short* arow_lo = xlo + (size_t)(mBase + col) * EMB_DIM;

    float best[8];
    int   bidx[8];
    #pragma unroll
    for (int r = 0; r < 8; r++) { best[r] = 3.4e38f; bidx[r] = 0; }

    // prologue: stage panel 0 into buffer 0
    stage_panel(ehi, elo, sbhi[0], sblo[0], t);

    for (int ct = 0; ct < N_EMBS / 16; ct++) {
        int buf = ct & 1;
        wait_async0();        // own panel-ct chunks landed in LDS
        __syncthreads();      // panel ct visible block-wide
        if (ct + 1 < N_EMBS / 16) {
            // buffer buf^1 was last read in iteration ct-1, before the barrier
            stage_panel(ehi + (size_t)(ct + 1) * PANEL,
                        elo + (size_t)(ct + 1) * PANEL,
                        sbhi[buf ^ 1], sblo[buf ^ 1], t);
        }
        const unsigned short* bh = &sbhi[buf][col * EMB_DIM + sel * 8];
        const unsigned short* bl = &sblo[buf][col * EMB_DIM + sel * 8];

        v8f acc = {};
        #pragma unroll
        for (int ks = 0; ks < 8; ks++) {   // 8 * 32 = 256 = EMB_DIM
            int kb = ks * 32;
            union { v16bf v; u32x4 q[2]; } ahi, alo, bhi, blo;
            ahi.q[0] = *(const u32x4*)(arow_hi + kb + sel * 8);
            ahi.q[1] = *(const u32x4*)(arow_hi + kb + 16 + sel * 8);
            alo.q[0] = *(const u32x4*)(arow_lo + kb + sel * 8);
            alo.q[1] = *(const u32x4*)(arow_lo + kb + 16 + sel * 8);
            bhi.q[0] = *(const u32x4*)(bh + kb);        // ds_load_b128
            bhi.q[1] = *(const u32x4*)(bh + kb + 16);
            blo.q[0] = *(const u32x4*)(bl + kb);
            blo.q[1] = *(const u32x4*)(bl + kb + 16);
            acc = __builtin_amdgcn_wmma_f32_16x16x32_bf16(false, ahi.v, false, bhi.v,
                                                          (short)0, acc, false, false);
            acc = __builtin_amdgcn_wmma_f32_16x16x32_bf16(false, ahi.v, false, blo.v,
                                                          (short)0, acc, false, false);
            acc = __builtin_amdgcn_wmma_f32_16x16x32_bf16(false, alo.v, false, bhi.v,
                                                          (short)0, acc, false, false);
        }
        int code = ct * 16 + col;
        float ev = e2[code];
        #pragma unroll
        for (int r = 0; r < 8; r++) {
            float s = ev - 2.0f * acc[r];
            if (s < best[r]) { best[r] = s; bidx[r] = code; }  // ascending: keeps first min
        }
    }
    // min-reduce across the 16 code columns within each half-wave
    #pragma unroll
    for (int off = 8; off >= 1; off >>= 1) {
        #pragma unroll
        for (int r = 0; r < 8; r++) {
            float ov = __shfl_xor(best[r], off, 16);
            int   oi = __shfl_xor(bidx[r], off, 16);
            if (ov < best[r] || (ov == best[r] && oi < bidx[r])) {
                best[r] = ov; bidx[r] = oi;
            }
        }
    }
    if (col == 0) {
        #pragma unroll
        for (int r = 0; r < 8; r++)
            bestIdx[mBase + sel * 8 + r] = bidx[r];   // lane0: rows 0..7, lane16: rows 8..15
    }
}

// ---------------------------------------------------------------------------
// Kernel 4: z_q gather (straight-through output value == emb[idx]) + SSE partials.
// ---------------------------------------------------------------------------
__global__ void k_zq(const float* __restrict__ x,
                     const float* __restrict__ emb,
                     const int* __restrict__ bestIdx,
                     float* __restrict__ zq_out,
                     float* __restrict__ partials) {
    int blk = blockIdx.x;
    int dt = blk & 7;
    int h  = (blk >> 3) & 31;
    int b  = blk >> 8;
    int tx = threadIdx.x & 31;   // w
    int ty = threadIdx.x >> 5;   // 0..7
    __shared__ int sidx[32];
    if (threadIdx.x < 32)
        sidx[threadIdx.x] = bestIdx[b * 1024 + h * 32 + threadIdx.x];
    __syncthreads();
    int idx = sidx[tx];
    float sse = 0.f;
    #pragma unroll
    for (int i = 0; i < 4; i++) {
        int d = dt * 32 + ty + i * 8;
        size_t go = (((size_t)b * EMB_DIM + d) * HH + h) * WW + tx;
        float xv = x[go];
        float e  = emb[(size_t)idx * EMB_DIM + d];   // hot in L2 (1 MB)
        float df = xv - e;
        sse += df * df;
        zq_out[go] = e;
    }
    __shared__ float red[256];
    red[threadIdx.x] = sse;
    __syncthreads();
    for (int s = 128; s > 0; s >>= 1) {
        if (threadIdx.x < s) red[threadIdx.x] += red[threadIdx.x + s];
        __syncthreads();
    }
    if (threadIdx.x == 0) partials[blk] = red[0];
}

// ---------------------------------------------------------------------------
// Kernel 5: deterministic fixed-order reduction of SSE partials -> vq_loss.
// ---------------------------------------------------------------------------
__global__ void k_loss(const float* __restrict__ partials, float* __restrict__ out0) {
    __shared__ float red[256];
    float s = 0.f;
    for (int i = threadIdx.x; i < 8192; i += 256) s += partials[i];
    red[threadIdx.x] = s;
    __syncthreads();
    for (int t = 128; t > 0; t >>= 1) {
        if (threadIdx.x < t) red[threadIdx.x] += red[threadIdx.x + t];
        __syncthreads();
    }
    if (threadIdx.x == 0)
        out0[0] = BETA * red[0] / (float)((size_t)BATCH * EMB_DIM * HH * WW);
}

// ---------------------------------------------------------------------------
// Kernel 6: LDS integer histogram (deterministic) + Laplace-smoothed cluster size.
// ---------------------------------------------------------------------------
__global__ void k_cs(const int* __restrict__ bestIdx,
                     const float* __restrict__ cluster_size,
                     float* __restrict__ cs_out) {
    __shared__ int   cnt[N_EMBS];
    __shared__ float red[N_EMBS];
    int t = threadIdx.x;
    cnt[t] = 0;
    __syncthreads();
    for (int n = t; n < NROWS; n += 1024)
        atomicAdd(&cnt[bestIdx[n]], 1);          // integer LDS atomic: deterministic
    __syncthreads();
    float cs = cluster_size[t] * DECAY + (float)cnt[t] * (1.0f - DECAY);
    red[t] = cs;
    __syncthreads();
    for (int s = 512; s > 0; s >>= 1) {
        if (t < s) red[t] += red[t + s];
        __syncthreads();
    }
    float n_tot = red[0];
    cs_out[t] = (cs + EPSV) / (n_tot + (float)N_EMBS * EPSV) * n_tot;
}

// ---------------------------------------------------------------------------
// Kernel 7: deterministic segment-sum (batch_sum) + EMA update -> emb_new.
// ---------------------------------------------------------------------------
__global__ void k_embnew(const unsigned short* __restrict__ xhi,
                         const unsigned short* __restrict__ xlo,
                         const int* __restrict__ bestIdx,
                         const float* __restrict__ ema_sum,
                         const float* __restrict__ cs,
                         float* __restrict__ emb_new) {
    int k = blockIdx.x, d = threadIdx.x;
    float acc = 0.f;
    for (int n = 0; n < NROWS; n++) {
        if (bestIdx[n] == k) {                   // uniform scalar branch
            size_t o = (size_t)n * EMB_DIM + d;  // coalesced bf16 reads
            acc += bf2f(xhi[o]) + bf2f(xlo[o]);
        }
    }
    float ema = ema_sum[(size_t)k * EMB_DIM + d] * DECAY + acc * (1.0f - DECAY);
    emb_new[(size_t)k * EMB_DIM + d] = ema / cs[k];
}

// ---------------------------------------------------------------------------
// Host launcher.
// d_in: x(8388608), emb_weight(262144), ema_sum(262144), cluster_size(1024)
// d_out: [vq_loss(1) | z_q_out(8388608) | emb_new(262144)]
// ws usage: ~33.3 MB.
// ---------------------------------------------------------------------------
extern "C" void kernel_launch(void* const* d_in, const int* in_sizes, int n_in,
                              void* d_out, int out_size, void* d_ws, size_t ws_size,
                              hipStream_t stream) {
    (void)in_sizes; (void)n_in; (void)out_size; (void)ws_size;
    const float* x   = (const float*)d_in[0];
    const float* emb = (const float*)d_in[1];
    const float* ema = (const float*)d_in[2];
    const float* csz = (const float*)d_in[3];
    float* out = (float*)d_out;

    char* w = (char*)d_ws;
    size_t off = 0;
    auto carve = [&](size_t bytes) -> void* {
        void* p = w + off;
        off += (bytes + 255) & ~(size_t)255;
        return p;
    };
    unsigned short* xhi = (unsigned short*)carve((size_t)NROWS * EMB_DIM * 2);   // 16 MB
    unsigned short* xlo = (unsigned short*)carve((size_t)NROWS * EMB_DIM * 2);   // 16 MB
    unsigned short* ehi = (unsigned short*)carve((size_t)N_EMBS * EMB_DIM * 2);  // 512 KB
    unsigned short* elo = (unsigned short*)carve((size_t)N_EMBS * EMB_DIM * 2);  // 512 KB
    float* e2       = (float*)carve(N_EMBS * sizeof(float));
    int*   bestIdx  = (int*)carve(NROWS * sizeof(int));
    float* partials = (float*)carve(8192 * sizeof(float));
    float* cs       = (float*)carve(N_EMBS * sizeof(float));

    float* zq_out  = out + 1;
    float* emb_new = out + 1 + (size_t)BATCH * EMB_DIM * HH * WW;

    k_prep_emb<<<N_EMBS, EMB_DIM, 0, stream>>>(emb, ehi, elo, e2);
    k_prep_x<<<BATCH * HH * (EMB_DIM / 32), 256, 0, stream>>>(x, xhi, xlo);
    k_argmin<<<NROWS / 128, 256, 0, stream>>>(xhi, xlo, ehi, elo, e2, bestIdx);
    k_zq<<<8192, 256, 0, stream>>>(x, emb, bestIdx, zq_out, partials);
    k_loss<<<1, 256, 0, stream>>>(partials, out);
    k_cs<<<1, 1024, 0, stream>>>(bestIdx, csz, cs);
    k_embnew<<<N_EMBS, EMB_DIM, 0, stream>>>(xhi, xlo, bestIdx, ema, cs, emb_new);
}
